// Net_37520834298136
// MI455X (gfx1250) — matrix operand order repaired and checked by reading, not compiled, not run on hardware
//
#include <hip/hip_runtime.h>
#include <hip/hip_bf16.h>
#include <math.h>

// ---- model dims (compile-time, from reference) ----
#define BATCH     4
#define SEQ       2048
#define TOKENS    (BATCH * SEQ)        // 8192
#define IN_DIM    64
#define D_MODEL   256
#define N_LAYERS  4
#define D_INNER   512                   // 2*D_MODEL
#define D_STATE   16
#define D_CONV    4
#define DT_RANK   16
#define XPROJ_N   (DT_RANK + 2 * D_STATE)  // 48

typedef __bf16 bf16;
typedef __attribute__((ext_vector_type(16))) __bf16 v16bf;
typedef __attribute__((ext_vector_type(8)))  __bf16 v8bf;
typedef __attribute__((ext_vector_type(8)))  float  v8f;

// ---------------------------------------------------------------------------
// f32 -> bf16 conversion (weights & x), elementwise
// ---------------------------------------------------------------------------
__global__ void cvt_f32_bf16(const float* __restrict__ in, bf16* __restrict__ out, int n) {
    int i = blockIdx.x * 256 + threadIdx.x;
    if (i < n) out[i] = (bf16)in[i];
}

// ---------------------------------------------------------------------------
// WMMA GEMM:  out[M,N] (f32) = A[M,K] (bf16 row-major) @ Bw[N,K]^T (bf16 row-major)
//             (+ bias[N]) (+ resid[M,N])
// One wave computes a 16 x (16*NT) output strip: the A fragment is loaded once
// per K-step and reused across NT accumulators (NT WMMAs per A load), raising
// arithmetic intensity 4x vs one-tile-per-wave. Operands load as aligned b128s
// directly from global per the CDNA5 WMMA VGPR layouts (A rows contiguous in K;
// B column n == weight row n, contiguous in K).
// Grid: total_waves/8 blocks of 256 threads (8 waves); all shapes used keep
// M%16==0, N%(16*NT)==0, K%32==0 and total_waves%8==0.
// ---------------------------------------------------------------------------
template <int NT>
__global__ void wmma_gemm_bf16(const bf16* __restrict__ A,
                               const bf16* __restrict__ Bw,
                               const float* __restrict__ bias,
                               const float* __restrict__ resid,
                               float* __restrict__ out,
                               int M, int N, int K) {
    const int lane    = threadIdx.x & 31;
    const int waveId  = blockIdx.x * 8 + (threadIdx.x >> 5);
    const int ngroups = N / (16 * NT);
    const int tile_m  = waveId / ngroups;
    const int tile_n0 = (waveId % ngroups) * NT;

    // A fragment addressing (16-bit A 16x32 layout)
    const int arow   = tile_m * 16 + (lane & 15);
    const int akbase = (lane >> 4) << 3;          // 0 or 8
    // B fragment addressing (16-bit B 32x16 layout); B column n == Bw row n
    const int bn     = tile_n0 * 16 + (lane & 15);
    const int bkbase = (lane >> 4) << 3;          // 0 or 8

    v8f acc[NT];
#pragma unroll
    for (int j = 0; j < NT; ++j) acc[j] = (v8f){};

    const bf16* ap = A + (size_t)arow * K + akbase;
    const bf16* bp[NT];
#pragma unroll
    for (int j = 0; j < NT; ++j) bp[j] = Bw + (size_t)(bn + 16 * j) * K + bkbase;

    for (int k0 = 0; k0 < K; k0 += 32) {
        __builtin_prefetch(ap + 32, 0, 1);
        v8bf alo = *(const v8bf*)(ap);            // K = k0+akbase .. +7
        v8bf ahi = *(const v8bf*)(ap + 16);       // K = k0+akbase+16 .. +23
        v16bf av;
#pragma unroll
        for (int i = 0; i < 8; ++i) { av[i] = alo[i]; av[8 + i] = ahi[i]; }
#pragma unroll
        for (int j = 0; j < NT; ++j) {
            __builtin_prefetch(bp[j] + 32, 0, 1);
            v8bf blo = *(const v8bf*)(bp[j]);     // K = k0+bkbase .. +7
            v8bf bhi = *(const v8bf*)(bp[j] + 16); // K = k0+bkbase+16 .. +23
            v16bf bv;
#pragma unroll
            for (int i = 0; i < 8; ++i) { bv[i] = blo[i]; bv[8 + i] = bhi[i]; }
            acc[j] = __builtin_amdgcn_wmma_f32_16x16x32_bf16(
                /*neg_a=*/false, av, /*neg_b=*/false, bv,
                /*c_mod=*/(short)0, acc[j], /*reuse_a=*/false, /*reuse_b=*/false);
            bp[j] += 32;
        }
        ap += 32;
    }

    // D layout: lane 0-15 -> N=lane, M=r ; lane 16-31 -> N=lane-16, M=r+8
    const int om0 = tile_m * 16 + ((lane >> 4) << 3);
#pragma unroll
    for (int j = 0; j < NT; ++j) {
        const int on = (tile_n0 + j) * 16 + (lane & 15);
        const float bv0 = bias ? bias[on] : 0.0f;
#pragma unroll
        for (int r = 0; r < 8; ++r) {
            const size_t idx = (size_t)(om0 + r) * N + on;
            float v = acc[j][r] + bv0;
            if (resid) v += resid[idx];
            out[idx] = v;
        }
    }
}

// ---------------------------------------------------------------------------
// RMSNorm over D_MODEL=256, output bf16 (feeds in_proj WMMA). One block/row.
// ---------------------------------------------------------------------------
__global__ void rmsnorm_bf16(const float* __restrict__ h,
                             const float* __restrict__ w,
                             bf16* __restrict__ out) {
    __shared__ float red[D_MODEL];
    const int row = blockIdx.x, c = threadIdx.x;
    const float v = h[(size_t)row * D_MODEL + c];
    red[c] = v * v;
    __syncthreads();
    for (int s = D_MODEL / 2; s > 0; s >>= 1) {
        if (c < s) red[c] += red[c + s];
        __syncthreads();
    }
    const float scale = rsqrtf(red[0] * (1.0f / D_MODEL) + 1e-5f);
    out[(size_t)row * D_MODEL + c] = (bf16)(v * scale * w[c]);
}

// ---------------------------------------------------------------------------
// Depthwise causal conv (width 4) + bias + SiLU. xz holds [xs|z] (width 1024).
// Emits xs in both f32 (for the scan) and bf16 (for the x_proj WMMA).
// ---------------------------------------------------------------------------
__global__ void conv_silu(const float* __restrict__ xz,
                          const float* __restrict__ cw,   // (D_INNER,4)
                          const float* __restrict__ cb,   // (D_INNER,)
                          float* __restrict__ xsf,
                          bf16* __restrict__ xsb) {
    const int idx = blockIdx.x * 256 + threadIdx.x;       // over TOKENS*D_INNER
    const int d   = idx & (D_INNER - 1);
    const int tok = idx >> 9;
    const int t   = tok & (SEQ - 1);
    float acc = cb[d];
#pragma unroll
    for (int j = 0; j < D_CONV; ++j) {
        const int tt = t - (D_CONV - 1 - j);
        if (tt >= 0)
            acc += cw[d * D_CONV + j] *
                   xz[(size_t)(tok - (D_CONV - 1 - j)) * (2 * D_INNER) + d];
    }
    const float s = acc / (1.0f + __expf(-acc));
    xsf[idx] = s;
    xsb[idx] = (bf16)s;
}

// ---------------------------------------------------------------------------
// delta = softplus(dbc[:, :DT_RANK] @ dt_w^T + dt_b)   (K=16 — scalar VALU)
// ---------------------------------------------------------------------------
__global__ void dt_softplus(const float* __restrict__ dbc,   // (TOKENS,48)
                            const float* __restrict__ dtw,   // (D_INNER,16)
                            const float* __restrict__ dtb,   // (D_INNER,)
                            float* __restrict__ delta) {
    const int idx = blockIdx.x * 256 + threadIdx.x;
    const int d   = idx & (D_INNER - 1);
    const int tok = idx >> 9;
    const float* r = dbc + (size_t)tok * XPROJ_N;
    float acc = dtb[d];
#pragma unroll
    for (int s = 0; s < DT_RANK; ++s) acc += r[s] * dtw[d * DT_RANK + s];
    delta[idx] = (acc > 20.0f) ? acc : log1pf(__expf(acc));
}

// ---------------------------------------------------------------------------
// Selective scan. One thread per (b, d): 16-state fp32 recurrence in regs,
// B/C broadcast-loaded (uniform address per wave). Fused D-skip + silu(z) gate.
// Grid: BATCH blocks x D_INNER threads.
// ---------------------------------------------------------------------------
__global__ void selective_scan(const float* __restrict__ xsf,
                               const float* __restrict__ delta,
                               const float* __restrict__ dbc,
                               const float* __restrict__ xz,
                               const float* __restrict__ A_log, // (D_INNER,16)
                               const float* __restrict__ Dp,    // (D_INNER,)
                               bf16* __restrict__ yb) {
    const int b = blockIdx.x;
    const int d = threadIdx.x;
    float Ar[D_STATE], hs[D_STATE];
#pragma unroll
    for (int s = 0; s < D_STATE; ++s) {
        Ar[s] = -__expf(A_log[d * D_STATE + s]);
        hs[s] = 0.0f;
    }
    const float Dpv = Dp[d];
    for (int t = 0; t < SEQ; ++t) {
        const size_t tok = (size_t)b * SEQ + t;
        const float dt = delta[tok * D_INNER + d];
        const float x  = xsf[tok * D_INNER + d];
        const float dx = dt * x;
        const float* bc = dbc + tok * XPROJ_N;
        float y = 0.0f;
#pragma unroll
        for (int s = 0; s < D_STATE; ++s) {
            const float dA = __expf(dt * Ar[s]);
            hs[s] = dA * hs[s] + dx * bc[DT_RANK + s];
            y += hs[s] * bc[DT_RANK + D_STATE + s];
        }
        y += Dpv * x;
        const float z = xz[tok * (2 * D_INNER) + D_INNER + d];
        y *= z / (1.0f + __expf(-z));
        yb[tok * D_INNER + d] = (bf16)y;
    }
}

// ---------------------------------------------------------------------------
// Head: sigmoid(h @ w_out^T + b_out)[...,0]. One thread per token.
// ---------------------------------------------------------------------------
__global__ void head_sigmoid(const float* __restrict__ h,
                             const float* __restrict__ wout,  // (1,D_MODEL)
                             const float* __restrict__ bout,  // (1,)
                             float* __restrict__ out) {
    const int tok = blockIdx.x * 256 + threadIdx.x;
    if (tok >= TOKENS) return;
    float acc = bout[0];
#pragma unroll 8
    for (int c = 0; c < D_MODEL; ++c) acc += h[(size_t)tok * D_MODEL + c] * wout[c];
    out[tok] = 1.0f / (1.0f + __expf(-acc));
}

// ---------------------------------------------------------------------------
extern "C" void kernel_launch(void* const* d_in, const int* in_sizes, int n_in,
                              void* d_out, int out_size, void* d_ws, size_t ws_size,
                              hipStream_t stream) {
    (void)in_sizes; (void)n_in; (void)out_size; (void)ws_size;
    const float* x       = (const float*)d_in[0];   // (B,L,64)
    const float* w_in    = (const float*)d_in[1];   // (256,64)
    const float* b_in    = (const float*)d_in[2];   // (256,)
    const float* w_out   = (const float*)d_in[3];   // (1,256)
    const float* b_out   = (const float*)d_in[4];   // (1,)
    const float* norm_w  = (const float*)d_in[5];   // (4,256)
    const float* in_w    = (const float*)d_in[6];   // (4,1024,256)
    const float* conv_w  = (const float*)d_in[7];   // (4,512,1,4)
    const float* conv_b  = (const float*)d_in[8];   // (4,512)
    const float* xproj_w = (const float*)d_in[9];   // (4,48,512)
    const float* dt_w    = (const float*)d_in[10];  // (4,512,16)
    const float* dt_b    = (const float*)d_in[11];  // (4,512)
    const float* A_log   = (const float*)d_in[12];  // (4,512,16)
    const float* Dp      = (const float*)d_in[13];  // (4,512)
    const float* out_w   = (const float*)d_in[14];  // (4,256,512)
    float* out = (float*)d_out;

    // ---- workspace layout ----
    char* ws = (char*)d_ws;
    size_t off = 0;
    auto take = [&](size_t bytes) { char* p = ws + off; off = (off + bytes + 255) & ~(size_t)255; return p; };
    float* h_f32  = (float*)take((size_t)TOKENS * D_MODEL * 4);
    bf16*  d_bf   = (bf16*) take((size_t)TOKENS * D_MODEL * 2);
    float* xz     = (float*)take((size_t)TOKENS * 2 * D_INNER * 4);
    float* xs_f   = (float*)take((size_t)TOKENS * D_INNER * 4);
    bf16*  xs_b   = (bf16*) take((size_t)TOKENS * D_INNER * 2);
    float* dbc    = (float*)take((size_t)TOKENS * XPROJ_N * 4);
    float* delta  = (float*)take((size_t)TOKENS * D_INNER * 4);
    bf16*  y_b    = (bf16*) take((size_t)TOKENS * D_INNER * 2);
    bf16*  x_b    = (bf16*) take((size_t)TOKENS * IN_DIM * 2);
    bf16*  w_in_b = (bf16*) take((size_t)D_MODEL * IN_DIM * 2);
    bf16*  in_w_b = (bf16*) take((size_t)N_LAYERS * 2 * D_INNER * D_MODEL * 2);
    bf16*  xp_w_b = (bf16*) take((size_t)N_LAYERS * XPROJ_N * D_INNER * 2);
    bf16*  ow_b   = (bf16*) take((size_t)N_LAYERS * D_MODEL * D_INNER * 2);

    auto cvt = [&](const float* src, bf16* dst, int n) {
        cvt_f32_bf16<<<(n + 255) / 256, 256, 0, stream>>>(src, dst, n);
    };
    cvt(x,       x_b,    TOKENS * IN_DIM);
    cvt(w_in,    w_in_b, D_MODEL * IN_DIM);
    cvt(in_w,    in_w_b, N_LAYERS * 2 * D_INNER * D_MODEL);
    cvt(xproj_w, xp_w_b, N_LAYERS * XPROJ_N * D_INNER);
    cvt(out_w,   ow_b,   N_LAYERS * D_MODEL * D_INNER);

    // NT=4 strip GEMM for N%64==0; NT=3 for x_proj (N=48).
    auto gemm4 = [&](const bf16* A, const bf16* Bw, const float* bias,
                     const float* resid, float* o, int M, int N, int K) {
        const int waves = (M / 16) * (N / 64);   // divisible by 8 for all uses
        wmma_gemm_bf16<4><<<waves / 8, 256, 0, stream>>>(A, Bw, bias, resid, o, M, N, K);
    };
    auto gemm3 = [&](const bf16* A, const bf16* Bw, const float* bias,
                     const float* resid, float* o, int M, int N, int K) {
        const int waves = (M / 16) * (N / 48);
        wmma_gemm_bf16<3><<<waves / 8, 256, 0, stream>>>(A, Bw, bias, resid, o, M, N, K);
    };

    // embed: h = x @ w_in^T + b_in   (M=8192, K=64, N=256)
    gemm4(x_b, w_in_b, b_in, nullptr, h_f32, TOKENS, D_MODEL, IN_DIM);

    const int NE = TOKENS * D_INNER;             // elementwise grid over (tok,d)
    for (int l = 0; l < N_LAYERS; ++l) {
        rmsnorm_bf16<<<TOKENS, D_MODEL, 0, stream>>>(h_f32, norm_w + l * D_MODEL, d_bf);
        // in_proj: xz = d @ in_w^T   (K=256, N=1024)
        gemm4(d_bf, in_w_b + (size_t)l * 2 * D_INNER * D_MODEL, nullptr, nullptr,
              xz, TOKENS, 2 * D_INNER, D_MODEL);
        conv_silu<<<NE / 256, 256, 0, stream>>>(
            xz, conv_w + (size_t)l * D_INNER * D_CONV, conv_b + (size_t)l * D_INNER,
            xs_f, xs_b);
        // x_proj: dbc = xs @ xproj_w^T   (K=512, N=48)
        gemm3(xs_b, xp_w_b + (size_t)l * XPROJ_N * D_INNER, nullptr, nullptr,
              dbc, TOKENS, XPROJ_N, D_INNER);
        dt_softplus<<<NE / 256, 256, 0, stream>>>(
            dbc, dt_w + (size_t)l * D_INNER * DT_RANK, dt_b + (size_t)l * D_INNER, delta);
        selective_scan<<<BATCH, D_INNER, 0, stream>>>(
            xs_f, delta, dbc, xz, A_log + (size_t)l * D_INNER * D_STATE,
            Dp + (size_t)l * D_INNER, y_b);
        // out_proj + residual: h += y @ out_w^T   (K=512, N=256)
        gemm4(y_b, ow_b + (size_t)l * D_MODEL * D_INNER, nullptr, h_f32,
              h_f32, TOKENS, D_MODEL, D_INNER);
    }

    head_sigmoid<<<TOKENS / 256, 256, 0, stream>>>(h_f32, w_out, b_out, out);
}
